// PointNetSetAbstractionMsg_51737176047776
// MI455X (gfx1250) — compile-verified, matrix operand order checked by reading, not run on hardware
//
#include <hip/hip_runtime.h>

// ---------------------------------------------------------------------------
// PointNet++ Set Abstraction (MSG) for gfx1250.
// bf16 WMMA (v_wmma_f32_16x16x32_bf16) GEMMs, fully shape-specialized:
// weights live in registers across M-tiles (grid-stride), f32 accumulation,
// fused per-channel BN statistics in the GEMM epilogue.
// ---------------------------------------------------------------------------

typedef __bf16 bf16_t;
typedef __attribute__((ext_vector_type(8)))  __bf16 v8bf;
typedef __attribute__((ext_vector_type(16))) __bf16 v16bf;
typedef __attribute__((ext_vector_type(8)))  float  v8f;

#define BATCH 16
#define NPTS  4096
#define SPTS  512
#define BS    (BATCH * SPTS)      // 8192
#define PTF   64                  // input point feature channels
#define CIN0  67
#define CIN0P 96                  // 67 padded up to multiple of 32
#define BN_EPS 1e-5f
#define OUT_XYZ_FLOATS (BS * 3)   // 24576
#define OUT_FEAT 320

__device__ __forceinline__ v16bf cat16(v8bf a, v8bf b) {
  return __builtin_shufflevector(a, b, 0, 1, 2, 3, 4, 5, 6, 7,
                                 8, 9, 10, 11, 12, 13, 14, 15);
}

// ---------------------------------------------------------------------------
// 1) Farthest point sampling: one block per batch.
// ---------------------------------------------------------------------------
__global__ __launch_bounds__(256)
void fps_kernel(const float* __restrict__ xyz, int* __restrict__ fidx,
                float* __restrict__ newxyz, float* __restrict__ out) {
  __shared__ float dist[NPTS];
  __shared__ float rv[256];
  __shared__ int   ri[256];
  __shared__ int   sfar;
  __shared__ float c[3];
  const int b = blockIdx.x, t = threadIdx.x;
  const float* bx = xyz + (long long)b * NPTS * 3;
  for (int i = t; i < NPTS; i += 256) dist[i] = 1e10f;
  if (t == 0) sfar = 0;
  __syncthreads();
  for (int it = 0; it < SPTS; ++it) {
    if (t == 0) {
      int far = sfar;
      fidx[b * SPTS + it] = far;
      float x = bx[far * 3 + 0], y = bx[far * 3 + 1], z = bx[far * 3 + 2];
      c[0] = x; c[1] = y; c[2] = z;
      long long o = ((long long)b * SPTS + it) * 3;
      newxyz[o] = x; newxyz[o + 1] = y; newxyz[o + 2] = z;
      out[o] = x;    out[o + 1] = y;    out[o + 2] = z;
    }
    __syncthreads();
    float bv = -1.0f; int bi = 0;
    for (int i = t; i < NPTS; i += 256) {
      float dx = bx[i * 3 + 0] - c[0];
      float dy = bx[i * 3 + 1] - c[1];
      float dz = bx[i * 3 + 2] - c[2];
      float d = dx * dx + dy * dy + dz * dz;
      float nd = fminf(dist[i], d);
      dist[i] = nd;
      if (nd > bv) { bv = nd; bi = i; }   // ascending i -> first-max tiebreak
    }
    rv[t] = bv; ri[t] = bi;
    __syncthreads();
    for (int off = 128; off > 0; off >>= 1) {
      if (t < off) {
        if (rv[t + off] > rv[t] ||
            (rv[t + off] == rv[t] && ri[t + off] < ri[t])) {
          rv[t] = rv[t + off]; ri[t] = ri[t + off];
        }
      }
      __syncthreads();
    }
    if (t == 0) sfar = ri[0];
    __syncthreads();
  }
}

// ---------------------------------------------------------------------------
// 2) Ball query + group + concat -> bf16 A0 rows [row, CIN0P].
//    One wave per (b, s). First-K-in-index-order, pad with first neighbor.
//    Row is assembled in registers and emitted as 12 x b128 stores.
// ---------------------------------------------------------------------------
__device__ inline void write_row(bf16_t* __restrict__ row,
                                 const float* __restrict__ p,
                                 const float* __restrict__ x,
                                 float cx, float cy, float cz) {
  float ext[3] = {x[0] - cx, x[1] - cy, x[2] - cz};
  #pragma unroll
  for (int ch = 0; ch < CIN0P / 8; ++ch) {     // 12 chunks of 8 channels
    v8bf v;
    #pragma unroll
    for (int e = 0; e < 8; ++e) {
      int c = ch * 8 + e;
      float f = (c < PTF) ? p[c] : (c < CIN0 ? ext[c - PTF] : 0.0f);
      v[e] = (bf16_t)f;
    }
    *(v8bf*)(row + ch * 8) = v;
  }
}

__global__ __launch_bounds__(256)
void group_kernel(const float* __restrict__ xyz, const float* __restrict__ pts,
                  const float* __restrict__ newxyz, bf16_t* __restrict__ A,
                  float r2, int K) {
  const int wave = (blockIdx.x * blockDim.x + threadIdx.x) >> 5;
  const int lane = threadIdx.x & 31;
  const int b = wave >> 9;          // / SPTS
  const float cx = newxyz[wave * 3 + 0];
  const float cy = newxyz[wave * 3 + 1];
  const float cz = newxyz[wave * 3 + 2];
  const float* bx = xyz + (long long)b * NPTS * 3;
  const float* bp = pts + (long long)b * NPTS * PTF;
  bf16_t* rowbase = A + (long long)wave * K * CIN0P;

  int cnt = 0, first = 0;
  for (int base = 0; base < NPTS && cnt < K; base += 32) {
    int j = base + lane;
    float dx = bx[j * 3 + 0] - cx;
    float dy = bx[j * 3 + 1] - cy;
    float dz = bx[j * 3 + 2] - cz;
    float d = dx * dx + dy * dy + dz * dz;
    bool in = (d <= r2);
    unsigned mask = (unsigned)__ballot(in);
    if (cnt == 0 && mask) first = base + __builtin_ctz(mask);
    int rank = __popc(mask & ((1u << lane) - 1u));
    if (in && (cnt + rank) < K)
      write_row(rowbase + (long long)(cnt + rank) * CIN0P,
                bp + (long long)j * PTF, bx + (long long)j * 3, cx, cy, cz);
    cnt += __popc(mask);
  }
  if (cnt > K) cnt = K;
  for (int slot = cnt + lane; slot < K; slot += 32)
    write_row(rowbase + (long long)slot * CIN0P,
              bp + (long long)first * PTF, bx + (long long)first * 3, cx, cy, cz);
}

// ---------------------------------------------------------------------------
// 3) Weight pack: f32 W[Cin,Cout] -> bf16 WT[Cout, CinPad] (K zero-padded).
// ---------------------------------------------------------------------------
__global__ void pack_w_kernel(const float* __restrict__ w, bf16_t* __restrict__ wt,
                              int Cin, int CinPad, int Cout) {
  int i = blockIdx.x * blockDim.x + threadIdx.x;
  if (i >= Cout * CinPad) return;
  int n = i / CinPad, k = i - n * CinPad;
  wt[i] = (k < Cin) ? (bf16_t)w[k * Cout + n] : (bf16_t)0.0f;
}

// ---------------------------------------------------------------------------
// 4) GEMM + bias + fused per-channel sum/sumsq.  Shape-specialized <NT, KT>:
//    COUT = NT*16 output cols, LDA = KT*32 input channels.
//    256-thread block = 8 waves; each wave grid-strides over 16-row M tiles.
//    NT*KT <= 12: B fragments register-resident (loaded from LDS once/wave).
//    Larger shapes: B fragments read per tile from LDS (fully unrolled).
// ---------------------------------------------------------------------------
template <int NT, int KT>
__global__ __launch_bounds__(256)
void gemm_bn_kernel(const bf16_t* __restrict__ A, const bf16_t* __restrict__ WT,
                    const float* __restrict__ bias, bf16_t* __restrict__ H,
                    float* __restrict__ csum, float* __restrict__ csumsq,
                    int Mtiles) {
  constexpr int LDA  = KT * 32;
  constexpr int COUT = NT * 16;
  constexpr bool REGB = (NT * KT) <= 12;

  __shared__ __align__(16) bf16_t bsh[COUT * LDA];
  {
    const uint4* src = (const uint4*)WT;
    uint4* dst = (uint4*)bsh;
    #pragma unroll 2
    for (int i = threadIdx.x; i < (COUT * LDA) / 8; i += 256) dst[i] = src[i];
  }
  __syncthreads();

  const int lane = threadIdx.x & 31;
  const int wv   = threadIdx.x >> 5;
  const int half = lane >> 4;     // K sub-range selector per ISA layout
  const int l15  = lane & 15;
  const int gwave  = blockIdx.x * 8 + wv;
  const int nwaves = gridDim.x * 8;

  // per-column bias, loaded once
  float biasv[NT];
  #pragma unroll
  for (int nt = 0; nt < NT; ++nt) biasv[nt] = bias[nt * 16 + l15];

  // B fragments: register-resident when small enough
  v16bf bfrag[REGB ? NT * KT : 1];
  if constexpr (REGB) {
    #pragma unroll
    for (int nt = 0; nt < NT; ++nt) {
      const bf16_t* brow = bsh + (nt * 16 + l15) * LDA + half * 8;
      #pragma unroll
      for (int k = 0; k < KT; ++k)
        bfrag[nt * KT + k] = cat16(*(const v8bf*)(brow + k * 32),
                                   *(const v8bf*)(brow + k * 32 + 16));
    }
  }

  for (int tile = gwave; tile < Mtiles; tile += nwaves) {
    const long long mrow0 = (long long)tile * 16;
    const bf16_t* arow = A + (mrow0 + l15) * LDA + half * 8;

    // batch-load all A fragments for this tile (single drain point)
    v16bf afrag[KT];
    #pragma unroll
    for (int k = 0; k < KT; ++k)
      afrag[k] = cat16(*(const v8bf*)(arow + k * 32),
                       *(const v8bf*)(arow + k * 32 + 16));

    v8f acc[NT];
    #pragma unroll
    for (int nt = 0; nt < NT; ++nt) {
      #pragma unroll
      for (int r = 0; r < 8; ++r) acc[nt][r] = biasv[nt];
    }

    #pragma unroll
    for (int nt = 0; nt < NT; ++nt) {
      #pragma unroll
      for (int k = 0; k < KT; ++k) {
        v16bf bv;
        if constexpr (REGB) {
          bv = bfrag[nt * KT + k];
        } else {
          const bf16_t* brow = bsh + (nt * 16 + l15) * LDA + half * 8 + k * 32;
          bv = cat16(*(const v8bf*)brow, *(const v8bf*)(brow + 16));
        }
        acc[nt] = __builtin_amdgcn_wmma_f32_16x16x32_bf16(
            false, afrag[k], false, bv, (short)0, acc[nt], false, false);
      }
    }

    #pragma unroll
    for (int nt = 0; nt < NT; ++nt) {
      const int col = nt * 16 + l15;
      float s = 0.0f, q = 0.0f;
      #pragma unroll
      for (int r = 0; r < 8; ++r) {
        float v = acc[nt][r];
        s += v; q += v * v;
        long long row = mrow0 + r + half * 8;   // C layout: lanes>=16 -> M+8
        H[row * COUT + col] = (bf16_t)v;
      }
      s += __shfl_xor(s, 16, 32);
      q += __shfl_xor(q, 16, 32);
      if (half == 0) {
        atomicAdd(&csum[col], s);
        atomicAdd(&csumsq[col], q);
      }
    }
  }
}

// ---------------------------------------------------------------------------
// 5) BN finalize: sums -> per-channel scale/shift.
// ---------------------------------------------------------------------------
__global__ void finalize_kernel(const float* __restrict__ csum,
                                const float* __restrict__ csumsq,
                                const float* __restrict__ g,
                                const float* __restrict__ t,
                                float* __restrict__ scale,
                                float* __restrict__ shift,
                                float invM, int Cout) {
  int c = threadIdx.x;
  if (c < Cout) {
    float mu  = csum[c] * invM;
    float var = csumsq[c] * invM - mu * mu;
    float sc  = g[c] * rsqrtf(var + BN_EPS);
    scale[c] = sc;
    shift[c] = t[c] - mu * sc;
  }
}

// ---------------------------------------------------------------------------
// 6) BN + ReLU, in place, vectorized 8 channels per thread (b128 traffic).
// ---------------------------------------------------------------------------
__global__ __launch_bounds__(256)
void bnrelu_kernel(bf16_t* __restrict__ H,
                   const float* __restrict__ scale,
                   const float* __restrict__ shift,
                   long long total8, int Cout) {
  __shared__ float ssc[128], ssh[128];
  if (threadIdx.x < 128) {
    ssc[threadIdx.x] = scale[threadIdx.x % Cout];
    ssh[threadIdx.x] = shift[threadIdx.x % Cout];
  }
  __syncthreads();
  long long i = (long long)blockIdx.x * blockDim.x + threadIdx.x;
  long long stride = (long long)gridDim.x * blockDim.x;
  for (; i < total8; i += stride) {
    int c0 = (int)((i * 8) % Cout);
    v8bf v = *(const v8bf*)(H + i * 8);
    v8bf o;
    #pragma unroll
    for (int e = 0; e < 8; ++e) {
      float y = (float)v[e] * ssc[c0 + e] + ssh[c0 + e];
      o[e] = (bf16_t)fmaxf(y, 0.0f);
    }
    *(v8bf*)(H + i * 8) = o;
  }
}

// ---------------------------------------------------------------------------
// 7) Max over K -> f32 features; 8 channels per thread.
// ---------------------------------------------------------------------------
__global__ __launch_bounds__(256)
void maxpool_kernel(const bf16_t* __restrict__ H, float* __restrict__ out,
                    int K, int Cout, int chanOff) {
  long long i = (long long)blockIdx.x * blockDim.x + threadIdx.x;
  long long total = (long long)BS * Cout / 8;
  if (i >= total) return;
  int c0 = (int)((i * 8) % Cout);
  long long bs = (i * 8) / Cout;
  const bf16_t* base = H + (bs * K) * Cout + c0;
  float m[8];
  #pragma unroll
  for (int e = 0; e < 8; ++e) m[e] = -1e30f;
  for (int k = 0; k < K; ++k) {
    v8bf v = *(const v8bf*)(base + (long long)k * Cout);
    #pragma unroll
    for (int e = 0; e < 8; ++e) m[e] = fmaxf(m[e], (float)v[e]);
  }
  float* o = out + OUT_XYZ_FLOATS + bs * OUT_FEAT + chanOff + c0;
  #pragma unroll
  for (int e = 0; e < 8; ++e) o[e] = m[e];
}

// ---------------------------------------------------------------------------
// Host orchestration
// ---------------------------------------------------------------------------
struct LayerCfg { int Cin, CinPad, Cout; };

static void launch_gemm(int NT, int KT, int Mtiles, const bf16_t* A,
                        const bf16_t* WT, const float* bias, bf16_t* H,
                        float* csum, float* csumsq, hipStream_t stream) {
  int blocks = Mtiles / 8;
  if (blocks > 4096) blocks = 4096;   // grid-stride -> B reuse across tiles
  #define GEMM_CASE(nt, kt)                                                   \
    if (NT == nt && KT == kt) {                                               \
      gemm_bn_kernel<nt, kt><<<blocks, 256, 0, stream>>>(A, WT, bias, H,      \
                                                         csum, csumsq, Mtiles);\
      return;                                                                 \
    }
  GEMM_CASE(2, 1) GEMM_CASE(2, 3) GEMM_CASE(4, 1) GEMM_CASE(4, 2)
  GEMM_CASE(4, 3) GEMM_CASE(6, 2) GEMM_CASE(8, 2) GEMM_CASE(8, 3)
  #undef GEMM_CASE
}

extern "C" void kernel_launch(void* const* d_in, const int* in_sizes, int n_in,
                              void* d_out, int out_size, void* d_ws, size_t ws_size,
                              hipStream_t stream) {
  (void)in_sizes; (void)n_in; (void)out_size; (void)ws_size;
  const float* xyz    = (const float*)d_in[0];
  const float* points = (const float*)d_in[1];
  auto Wp = [&](int i, int j) { return (const float*)d_in[2 + (i * 3 + j) * 4 + 0]; };
  auto Cp = [&](int i, int j) { return (const float*)d_in[2 + (i * 3 + j) * 4 + 1]; };
  auto Gp = [&](int i, int j) { return (const float*)d_in[2 + (i * 3 + j) * 4 + 2]; };
  auto Tp = [&](int i, int j) { return (const float*)d_in[2 + (i * 3 + j) * 4 + 3]; };
  float* out = (float*)d_out;

  // workspace carve-up
  char* wsb = (char*)d_ws;
  size_t off = 0;
  auto take = [&](size_t bytes) -> void* {
    void* p = wsb + off;
    off = (off + bytes + 255) & ~(size_t)255;
    return p;
  };
  int*    fidx   = (int*)take((size_t)BS * 4);
  float*  newxyz = (float*)take((size_t)BS * 3 * 4);
  float*  stats  = (float*)take(256 * 4);          // csum[128] ++ csumsq[128]
  float*  csum   = stats;
  float*  csumsq = stats + 128;
  float*  scale  = (float*)take(128 * 4);
  float*  shift  = (float*)take(128 * 4);
  bf16_t* wpack  = (bf16_t*)take((size_t)128 * 128 * 2);
  const size_t MMAX = (size_t)BS * 128;            // largest row count (K=128)
  bf16_t* ping   = (bf16_t*)take(MMAX * 128 * 2);
  bf16_t* pong   = (bf16_t*)take(MMAX * 128 * 2);

  // 1) FPS + new_xyz (also written to output)
  fps_kernel<<<BATCH, 256, 0, stream>>>(xyz, fidx, newxyz, out);

  static const float RAD[3] = {0.1f, 0.2f, 0.4f};
  static const int   KS[3]  = {16, 32, 128};
  static const int   CHOFF[3] = {0, 64, 192};
  static const LayerCfg CFG[3][3] = {
    {{CIN0, CIN0P, 32}, {32, 32, 32}, {32, 32, 64}},
    {{CIN0, CIN0P, 64}, {64, 64, 64}, {64, 64, 128}},
    {{CIN0, CIN0P, 64}, {64, 64, 96}, {96, 96, 128}},
  };

  for (int br = 0; br < 3; ++br) {
    const int K = KS[br];
    const long long M = (long long)BS * K;

    // 2) ball query + group -> ping (lda = CIN0P)
    group_kernel<<<BS / 8, 256, 0, stream>>>(xyz, points, newxyz, ping,
                                             RAD[br] * RAD[br], K);

    bf16_t* cur = ping;
    bf16_t* nxt = pong;
    for (int j = 0; j < 3; ++j) {
      const LayerCfg& L = CFG[br][j];
      const int NT = L.Cout / 16;
      const int KT = L.CinPad / 32;
      // pack weights (f32 -> bf16 transposed, K-padded)
      int wn = L.Cout * L.CinPad;
      pack_w_kernel<<<(wn + 255) / 256, 256, 0, stream>>>(Wp(br, j), wpack,
                                                          L.Cin, L.CinPad, L.Cout);
      // zero BN stat accumulators
      hipMemsetAsync(stats, 0, 256 * 4, stream);
      // GEMM + bias + fused channel sums (register/LDS-resident weights)
      launch_gemm(NT, KT, (int)(M / 16), cur, wpack, Cp(br, j), nxt,
                  csum, csumsq, stream);
      // BN scale/shift
      finalize_kernel<<<1, 128, 0, stream>>>(csum, csumsq, Gp(br, j), Tp(br, j),
                                             scale, shift, 1.0f / (float)M, L.Cout);
      // BN + ReLU in place (vectorized)
      long long total8 = M * L.Cout / 8;
      int blocks = (int)(((total8 + 255) / 256) < 4096 ? ((total8 + 255) / 256) : 4096);
      bnrelu_kernel<<<blocks, 256, 0, stream>>>(nxt, scale, shift, total8, L.Cout);
      // swap
      bf16_t* t = cur; cur = nxt; nxt = t;
    }

    // 7) max over K -> output features
    const int CoutF = CFG[br][2].Cout;
    long long ptotal = (long long)BS * CoutF / 8;
    maxpool_kernel<<<(int)((ptotal + 255) / 256), 256, 0, stream>>>(
        cur, out, K, CoutF, CHOFF[br]);
  }
}
